// InverseNet_4664334483666
// MI455X (gfx1250) — matrix operand order repaired and checked by reading, not compiled, not run on hardware
//
#include <hip/hip_runtime.h>
#include <hip/hip_bf16.h>

typedef __attribute__((ext_vector_type(16))) _Float16 v16h;
typedef __attribute__((ext_vector_type(8)))  _Float16 v8h;
typedef __attribute__((ext_vector_type(8)))  float    v8f;

#define BATCH 16384
#define HDIM  256
#define DKDIM 64
#define NSLOT 8

// ---------------------------------------------------------------------------
// Weight packing: B-fragment (32x16 f16) per CDNA5 WMMA layout.
// Fragment (kt,nt): lane l holds column N = nt*16 + (l&15); its 16 halves e=0..15
// hold K = kt*32 + (e>>3)*16 + (l>>4)*8 + (e&7).  Stored as wp[frag*512 + lane*16 + e] so the
// GEMM loads one coalesced 32-byte v16h per lane.
// ---------------------------------------------------------------------------
__global__ void pack_w_kernel(const float* __restrict__ w, _Float16* __restrict__ wp,
                              int K, int N, int Kpad) {
  int tid = blockIdx.x * blockDim.x + threadIdx.x;
  int total = Kpad * N;
  if (tid >= total) return;
  int frag   = tid >> 9;       // /512
  int within = tid & 511;
  int lane   = within >> 4;
  int e      = within & 15;
  int ntiles = N >> 4;
  int kt = frag / ntiles, nt = frag % ntiles;
  int kg = kt * 32 + ((e >> 3) << 4) + ((lane >> 4) << 3) + (e & 7);
  int ng = nt * 16 + (lane & 15);
  float val = (kg < K) ? w[kg * N + ng] : 0.0f;
  wp[tid] = (_Float16)val;
}

// A-fragment (16x32 f16) gather from row-major LDS (pitch in halves).
// lanes 0-15: row M=lane, K = e(0..7) then 16+e ; lanes 16-31: K = 8+e then 24+e.
__device__ inline v16h load_a_frag(const _Float16* s, int pitch) {
  int lane = threadIdx.x & 31;
  int m = lane & 15, g = lane >> 4;
  v16h a;
#pragma unroll
  for (int e = 0; e < 16; ++e) {
    int k = ((e >> 3) << 4) + (e & 7) + (g << 3);
    a[e] = s[m * pitch + k];
  }
  return a;
}

__device__ inline v8f wmma16(v16h a, v16h b, v8f c) {
  return __builtin_amdgcn_wmma_f32_16x16x32_f16(false, a, false, b, (short)0, c,
                                                false, false);
}

__device__ inline v8f zero8() {
  v8f z = {0.f, 0.f, 0.f, 0.f, 0.f, 0.f, 0.f, 0.f};
  return z;
}

// ---------------------------------------------------------------------------
// Fused 2-layer MLP: h2 = relu(relu(x@w1+b1)@w2+b2) -> obs (f16), row remapped
// to (batch*8 + slot_base + row%tpb).  128 rows / block, 8 waves, each wave one
// 16-row m-tile across all 16 n-tiles of H=256.
// ---------------------------------------------------------------------------
__global__ __launch_bounds__(256)
void mlp2_kernel(const float* __restrict__ x, int in_dim, int kt1,
                 const _Float16* __restrict__ w1p, const float* __restrict__ b1,
                 const _Float16* __restrict__ w2p, const float* __restrict__ b2,
                 _Float16* __restrict__ obs, int tpb, int slot_base) {
  __shared__ _Float16 xs[128][64];
  __shared__ _Float16 h1[128][264];
  __shared__ _Float16 h2[128][264];

  const int tid  = threadIdx.x;
  const int lane = tid & 31;
  const int wave = tid >> 5;
  const int row0 = blockIdx.x * 128;

  // Stage x tile as f16, zero-padded to 64 K-columns.
  for (int i = tid; i < 128 * 64; i += 256) {
    int r = i >> 6, c = i & 63;
    float v = (c < in_dim) ? x[(row0 + r) * in_dim + c] : 0.0f;
    xs[r][c] = (_Float16)v;
  }
  __syncthreads();

  const int g = lane >> 4, nn = lane & 15;

  // ---- layer 1 ----
  {
    v8f acc[16];
#pragma unroll
    for (int nt = 0; nt < 16; ++nt) acc[nt] = zero8();
    for (int kt = 0; kt < kt1; ++kt) {
      v16h a = load_a_frag(&xs[wave * 16][0] + kt * 32, 64);
#pragma unroll
      for (int nt = 0; nt < 16; ++nt) {
        v16h b = ((const v16h*)w1p)[(kt * 16 + nt) * 32 + lane];
        acc[nt] = wmma16(a, b, acc[nt]);
      }
    }
#pragma unroll
    for (int nt = 0; nt < 16; ++nt) {
      int col = nt * 16 + nn;
      float bias = b1[col];
#pragma unroll
      for (int r = 0; r < 8; ++r) {
        float v = acc[nt][r] + bias;
        v = v > 0.f ? v : 0.f;
        h1[wave * 16 + r + 8 * g][col] = (_Float16)v;
      }
    }
  }
  __syncthreads();

  // ---- layer 2 (K = 256 -> 8 k-steps) ----
  {
    v8f acc[16];
#pragma unroll
    for (int nt = 0; nt < 16; ++nt) acc[nt] = zero8();
    for (int kt = 0; kt < 8; ++kt) {
      v16h a = load_a_frag(&h1[wave * 16][0] + kt * 32, 264);
#pragma unroll
      for (int nt = 0; nt < 16; ++nt) {
        v16h b = ((const v16h*)w2p)[(kt * 16 + nt) * 32 + lane];
        acc[nt] = wmma16(a, b, acc[nt]);
      }
    }
#pragma unroll
    for (int nt = 0; nt < 16; ++nt) {
      int col = nt * 16 + nn;
      float bias = b2[col];
#pragma unroll
      for (int r = 0; r < 8; ++r) {
        float v = acc[nt][r] + bias;
        v = v > 0.f ? v : 0.f;
        h2[wave * 16 + r + 8 * g][col] = (_Float16)v;
      }
    }
  }
  __syncthreads();

  // Coalesced f16 store to obs with batch/slot row remap (16B chunks).
  for (int i = tid; i < 128 * 32; i += 256) {
    int r = i >> 5, c = i & 31;
    int grow  = row0 + r;
    int batch = grow / tpb;
    int slot  = slot_base + grow % tpb;
    int orow  = batch * NSLOT + slot;
    *(v8h*)(obs + orow * HDIM + c * 8) = *(const v8h*)(&h2[r][c * 8]);
  }
}

// ---------------------------------------------------------------------------
// q/k/v projection + fused attention.  One block = 16 batches (128 token rows).
// ---------------------------------------------------------------------------
__global__ __launch_bounds__(256)
void qkv_attn_kernel(const _Float16* __restrict__ obs,
                     const _Float16* __restrict__ qwp, const float* __restrict__ qb,
                     const _Float16* __restrict__ kwp, const float* __restrict__ kb,
                     const _Float16* __restrict__ vwp, const float* __restrict__ vb,
                     float* __restrict__ out_w, float* __restrict__ out_g,
                     float* __restrict__ out_v) {
  __shared__ _Float16 os[128][264];
  __shared__ _Float16 vs[128][264];
  __shared__ float    ks[128][68];
  __shared__ float    qs[16][64];
  __shared__ float    sc[16][8];
  __shared__ float    wt[16][8];

  const int tid  = threadIdx.x;
  const int lane = tid & 31;
  const int wave = tid >> 5;
  const int row0   = blockIdx.x * 128;  // token rows
  const int batch0 = blockIdx.x * 16;

  for (int i = tid; i < 128 * 32; i += 256) {
    int r = i >> 5, c = i & 31;
    *(v8h*)(&os[r][c * 8]) = *(const v8h*)(obs + (row0 + r) * HDIM + c * 8);
  }
  __syncthreads();

  const int g = lane >> 4, nn = lane & 15;

  // ---- V = relu-free affine: os @ vw + vb (fp32 out + f16 LDS copy) ----
  {
    v8f acc[16];
#pragma unroll
    for (int nt = 0; nt < 16; ++nt) acc[nt] = zero8();
    for (int kt = 0; kt < 8; ++kt) {
      v16h a = load_a_frag(&os[wave * 16][0] + kt * 32, 264);
#pragma unroll
      for (int nt = 0; nt < 16; ++nt) {
        v16h b = ((const v16h*)vwp)[(kt * 16 + nt) * 32 + lane];
        acc[nt] = wmma16(a, b, acc[nt]);
      }
    }
#pragma unroll
    for (int nt = 0; nt < 16; ++nt) {
      int col = nt * 16 + nn;
      float bias = vb[col];
#pragma unroll
      for (int r = 0; r < 8; ++r) {
        int lr = wave * 16 + r + 8 * g;
        float v = acc[nt][r] + bias;
        out_v[(row0 + lr) * HDIM + col] = v;
        vs[lr][col] = (_Float16)v;
      }
    }
  }

  // ---- Q and K (DK = 64 -> 4 n-tiles each) ----
  {
    v8f qa[4], ka[4];
#pragma unroll
    for (int nt = 0; nt < 4; ++nt) { qa[nt] = zero8(); ka[nt] = zero8(); }
    for (int kt = 0; kt < 8; ++kt) {
      v16h a = load_a_frag(&os[wave * 16][0] + kt * 32, 264);
#pragma unroll
      for (int nt = 0; nt < 4; ++nt) {
        v16h bq = ((const v16h*)qwp)[(kt * 4 + nt) * 32 + lane];
        qa[nt] = wmma16(a, bq, qa[nt]);
        v16h bk = ((const v16h*)kwp)[(kt * 4 + nt) * 32 + lane];
        ka[nt] = wmma16(a, bk, ka[nt]);
      }
    }
#pragma unroll
    for (int nt = 0; nt < 4; ++nt) {
      int col = nt * 16 + nn;
      float bqv = qb[col], bkv = kb[col];
#pragma unroll
      for (int r = 0; r < 8; ++r) {
        int lr = wave * 16 + r + 8 * g;
        ks[lr][col] = ka[nt][r] + bkv;
        if ((lr & 7) == 0) qs[lr >> 3][col] = qa[nt][r] + bqv;  // only token 0 q needed
      }
    }
  }
  __syncthreads();

  // ---- scores for q-row 0 of each batch ----
  if (tid < 128) {
    int bl = tid >> 3, j = tid & 7;
    float s = 0.f;
#pragma unroll
    for (int d = 0; d < DKDIM; ++d) s += qs[bl][d] * ks[bl * 8 + j][d];
    sc[bl][j] = s * 0.125f;  // 1/sqrt(64)
  }
  __syncthreads();

  // ---- softmax + weight output ----
  if (tid < 128) {
    int bl = tid >> 3, j = tid & 7;
    float mx = sc[bl][0];
#pragma unroll
    for (int t = 1; t < 8; ++t) mx = fmaxf(mx, sc[bl][t]);
    float den = 0.f;
#pragma unroll
    for (int t = 0; t < 8; ++t) den += __expf(sc[bl][t] - mx);
    float w = __expf(sc[bl][j] - mx) / den;
    wt[bl][j] = w;
    out_w[(batch0 + bl) * NSLOT + j] = w;
  }
  __syncthreads();

  // ---- weighted_gf[:,0,:] = weight @ v ----
  for (int i = tid; i < 16 * HDIM; i += 256) {
    int bl = i >> 8, h = i & 255;
    float a = 0.f;
#pragma unroll
    for (int j = 0; j < 8; ++j) a += wt[bl][j] * (float)vs[bl * 8 + j][h];
    out_g[(batch0 + bl) * HDIM + h] = a;
  }
}

// ---------------------------------------------------------------------------
extern "C" void kernel_launch(void* const* d_in, const int* in_sizes, int n_in,
                              void* d_out, int out_size, void* d_ws, size_t ws_size,
                              hipStream_t stream) {
  const float* action = (const float*)d_in[0];
  const float* wolf   = (const float*)d_in[1];
  const float* sheep  = (const float*)d_in[2];
  const float* wall   = (const float*)d_in[3];
  const float* aw1 = (const float*)d_in[4];  const float* ab1 = (const float*)d_in[5];
  const float* aw2 = (const float*)d_in[6];  const float* ab2 = (const float*)d_in[7];
  const float* ww1 = (const float*)d_in[8];  const float* wb1 = (const float*)d_in[9];
  const float* ww2 = (const float*)d_in[10]; const float* wb2 = (const float*)d_in[11];
  const float* sw1 = (const float*)d_in[12]; const float* sb1 = (const float*)d_in[13];
  const float* sw2 = (const float*)d_in[14]; const float* sb2 = (const float*)d_in[15];
  const float* lw1 = (const float*)d_in[16]; const float* lb1 = (const float*)d_in[17];
  const float* lw2 = (const float*)d_in[18]; const float* lb2 = (const float*)d_in[19];
  const float* qw  = (const float*)d_in[20]; const float* qb  = (const float*)d_in[21];
  const float* kw  = (const float*)d_in[22]; const float* kb  = (const float*)d_in[23];
  const float* vw  = (const float*)d_in[24]; const float* vb  = (const float*)d_in[25];

  // Workspace layout (f16 elements)
  _Float16* ws   = (_Float16*)d_ws;
  _Float16* aw1p = ws;               // 32*256
  _Float16* aw2p = aw1p + 32 * 256;  // 256*256
  _Float16* ww1p = aw2p + 256 * 256; // 64*256
  _Float16* ww2p = ww1p + 64 * 256;  // 256*256
  _Float16* sw1p = ww2p + 256 * 256;
  _Float16* sw2p = sw1p + 64 * 256;
  _Float16* lw1p = sw2p + 256 * 256;
  _Float16* lw2p = lw1p + 64 * 256;
  _Float16* qwp  = lw2p + 256 * 256; // 256*64
  _Float16* kwp  = qwp + 256 * 64;
  _Float16* vwp  = kwp + 256 * 64;   // 256*256
  _Float16* obsb = vwp + 256 * 256;  // BATCH*8*256 f16

  auto pack = [&](const float* w, _Float16* wp, int K, int N, int Kpad) {
    int total = Kpad * N;
    pack_w_kernel<<<dim3((total + 255) / 256), dim3(256), 0, stream>>>(w, wp, K, N, Kpad);
  };
  pack(aw1, aw1p, 5,   HDIM, 32);
  pack(aw2, aw2p, 256, HDIM, 256);
  pack(ww1, ww1p, 64,  HDIM, 64);
  pack(ww2, ww2p, 256, HDIM, 256);
  pack(sw1, sw1p, 64,  HDIM, 64);
  pack(sw2, sw2p, 256, HDIM, 256);
  pack(lw1, lw1p, 64,  HDIM, 64);
  pack(lw2, lw2p, 256, HDIM, 256);
  pack(qw,  qwp,  256, DKDIM, 256);
  pack(kw,  kwp,  256, DKDIM, 256);
  pack(vw,  vwp,  256, HDIM, 256);

  // Branch MLPs -> obs[B,8,256] f16
  mlp2_kernel<<<dim3(BATCH / 128),     dim3(256), 0, stream>>>(action, 5, 1, aw1p, ab1, aw2p, ab2, obsb, 1, 0);
  mlp2_kernel<<<dim3(BATCH * 3 / 128), dim3(256), 0, stream>>>(wolf,  64, 2, ww1p, wb1, ww2p, wb2, obsb, 3, 1);
  mlp2_kernel<<<dim3(BATCH * 3 / 128), dim3(256), 0, stream>>>(sheep, 64, 2, sw1p, sb1, sw2p, sb2, obsb, 3, 4);
  mlp2_kernel<<<dim3(BATCH / 128),     dim3(256), 0, stream>>>(wall,  64, 2, lw1p, lb1, lw2p, lb2, obsb, 1, 7);

  // Outputs: weight[:,0,:] (B*8) | weighted_gf[:,0,:] (B*256) | v (B*8*256)
  float* out_w = (float*)d_out;
  float* out_g = out_w + (size_t)BATCH * NSLOT;
  float* out_v = out_g + (size_t)BATCH * HDIM;

  qkv_attn_kernel<<<dim3(BATCH / 16), dim3(256), 0, stream>>>(obsb, qwp, qb, kwp, kb,
                                                              vwp, vb, out_w, out_g, out_v);
}
